// BiMambaBlock_40802189312121
// MI455X (gfx1250) — matrix operand order repaired and checked by reading, not compiled, not run on hardware
//
#include <hip/hip_runtime.h>

// ---------------------------------------------------------------------------
// Bi-directional Mamba block for gfx1250 (MI455X).
// GEMMs: v_wmma_f32_16x16x32_bf16, 128x128x32 tiles, double-buffered LDS,
// async global->LDS staging for the A tile. Scan/conv/norm in fp32.
// ---------------------------------------------------------------------------

typedef __bf16 bh;
typedef bh    v16bf __attribute__((ext_vector_type(16)));
typedef bh    v8bf  __attribute__((ext_vector_type(8)));
typedef float v8f   __attribute__((ext_vector_type(8)));

#define BB   2
#define LL   4096
#define DM   1024
#define DI   2048
#define DS   16
#define DR   64
#define XD   96            // DT_RANK + 2*D_STATE
#define ROWS (BB * LL)     // 8192
#define EPSF 1e-5f

// Flip to 0 if the toolchain rejects the async-to-LDS mnemonics.
#define USE_ASYNC_LDS 1

// ---------------------------- elementwise helpers --------------------------

__global__ __launch_bounds__(256) void k_f2bf(const float* __restrict__ s,
                                              bh* __restrict__ d, int n) {
  int i = blockIdx.x * 256 + threadIdx.x;
  if (i < n) d[i] = (bh)s[i];
}

__global__ __launch_bounds__(256) void k_rmsnorm(const float* __restrict__ x,
                                                 const float* __restrict__ nw,
                                                 bh* __restrict__ xn) {
  int r = blockIdx.x;
  const float* xr = x + (size_t)r * DM;
  float s = 0.f;
  for (int i = threadIdx.x; i < DM; i += 256) { float v = xr[i]; s += v * v; }
#pragma unroll
  for (int o = 16; o > 0; o >>= 1) s += __shfl_xor(s, o);
  __shared__ float red[9];
  if ((threadIdx.x & 31) == 0) red[threadIdx.x >> 5] = s;
  __syncthreads();
  if (threadIdx.x == 0) {
    float t = 0.f;
#pragma unroll
    for (int i = 0; i < 8; i++) t += red[i];
    red[8] = rsqrtf(t / (float)DM + EPSF);
  }
  __syncthreads();
  float sc = red[8];
  bh* o = xn + (size_t)r * DM;
  for (int i = threadIdx.x; i < DM; i += 256) o[i] = (bh)(xr[i] * sc * nw[i]);
}

// ------------------------------- WMMA GEMM ---------------------------------
// C[M,N] = epilogue( A[M,K](bf16, row stride lda) * W[N,K](bf16)^T )
// 256 threads / 8 waves; block tile 128x128x32; each wave owns 64x32 (8 WMMA).

#define TM 128
#define TN 128
#define TK 32
#define ASTR 40    // halves per A-tile row (32 + 8 pad)
#define WSTR 136   // halves per transposed-W k-row (128 + 8 pad)
#define ABUF (TM * ASTR)   // halves per A buffer
#define WBUF (TK * WSTR)   // halves per W buffer

enum { EPI_F32 = 0, EPI_SPLIT_SILU = 1, EPI_BF16 = 2, EPI_SOFTPLUS = 3,
       EPI_RESID = 4 };

__global__ __launch_bounds__(256) void k_gemm(
    const bh* __restrict__ A, int lda, const bh* __restrict__ W,
    int M, int N, int K, int mode,
    float* __restrict__ out0, bh* __restrict__ out1,
    const float* __restrict__ bias, const float* __restrict__ resid) {
  __shared__ __align__(16) bh a_sh[2 * ABUF];
  __shared__ __align__(16) bh w_sh[2 * WBUF];

  const int tid  = threadIdx.x;
  const int lane = tid & 31, wave = tid >> 5;
  const int hi   = lane >> 4, lm = lane & 15;
  const int bm = blockIdx.y * TM;
  const int bn = blockIdx.x * TN;
  const int wm = (wave >> 2) * 64;   // 2 wave-rows * 64 rows
  const int wn = (wave & 3) * 32;    // 4 wave-cols * 32 cols

  v8f acc[4][2];
#pragma unroll
  for (int a = 0; a < 4; a++)
#pragma unroll
    for (int c = 0; c < 2; c++)
#pragma unroll
      for (int j = 0; j < 8; j++) acc[a][c][j] = 0.f;

  const int frow = tid >> 1;         // 0..127
  const int fk   = (tid & 1) * 16;   // 0 / 16

#if USE_ASYNC_LDS
  // LDS byte offset of a_sh for the async DMA (generic -> AS3 -> u32).
  const unsigned a_base =
      (unsigned)(size_t)(__attribute__((address_space(3))) char*)(void*)a_sh;
#endif

  // ---- tile staging (A async -> LDS, W synchronous transpose scatter) ----
  auto stage = [&](int buf, int k0) {
    int gr = bm + frow;
    if (gr < M) {
      const bh* pA = A + (size_t)gr * lda + k0 + fk;
#if USE_ASYNC_LDS
      unsigned ao = a_base + (unsigned)(buf * ABUF + frow * ASTR + fk) * 2u;
      asm volatile("global_load_async_to_lds_b128 %0, %1, off"
                   :: "v"(ao), "v"(pA) : "memory");
      asm volatile("global_load_async_to_lds_b128 %0, %1, off"
                   :: "v"(ao + 16u), "v"(pA + 8) : "memory");
#else
      v8bf a0 = *(const v8bf*)pA;
      v8bf a1 = *(const v8bf*)(pA + 8);
      *(v8bf*)&a_sh[buf * ABUF + frow * ASTR + fk]     = a0;
      *(v8bf*)&a_sh[buf * ABUF + frow * ASTR + fk + 8] = a1;
#endif
    }
    int gn = bn + frow;
    if (gn < N) {
      const bh* pW = W + (size_t)gn * K + k0 + fk;
      v8bf w0 = *(const v8bf*)pW;
      v8bf w1 = *(const v8bf*)(pW + 8);
      if (k0 + TK < K) __builtin_prefetch(pW + TK, 0, 3);
      bh* wd = w_sh + buf * WBUF;
#pragma unroll
      for (int i = 0; i < 8; i++) {
        wd[(fk + i) * WSTR + frow]     = w0[i];
        wd[(fk + 8 + i) * WSTR + frow] = w1[i];
      }
    }
  };

  stage(0, 0);

  const int ksteps = K / TK;
  for (int kt = 0; kt < ksteps; kt++) {
    const int cur = kt & 1;
#if USE_ASYNC_LDS
    asm volatile("s_wait_asynccnt 0" ::: "memory");
#endif
    __syncthreads();
    if (kt + 1 < ksteps) stage(cur ^ 1, (kt + 1) * TK);

    const bh* ab = a_sh + cur * ABUF;
    const bh* wb = w_sh + cur * WBUF;

    // B fragments: lane l holds B[k, n=l%16], elem i: k = (i<8 ? 8*hi+i : 16+8*hi+(i-8))
    v16bf bfrag[2];
#pragma unroll
    for (int nt = 0; nt < 2; nt++) {
      const bh* p = &wb[wn + nt * 16 + lm];
#pragma unroll
      for (int i = 0; i < 8; i++) { bfrag[nt][i] = p[(8 * hi + i) * WSTR]; bfrag[nt][i + 8] = p[(16 + 8 * hi + i) * WSTR]; }
    }
#pragma unroll
    for (int mt = 0; mt < 4; mt++) {
      // A fragment, elem i: k = (i<8 ? 8*hi+i : 16+8*hi+(i-8))
      int ar = wm + mt * 16 + lm;
      v8bf lo = *(const v8bf*)&ab[ar * ASTR + 8 * hi];
      v8bf h2 = *(const v8bf*)&ab[ar * ASTR + 16 + 8 * hi];
      v16bf afrag;
#pragma unroll
      for (int i = 0; i < 8; i++) { afrag[i] = lo[i]; afrag[i + 8] = h2[i]; }
#pragma unroll
      for (int nt = 0; nt < 2; nt++)
        acc[mt][nt] = __builtin_amdgcn_wmma_f32_16x16x32_bf16(
            false, afrag, false, bfrag[nt], (short)0, acc[mt][nt],
            false, false);
    }
  }

  // ---- epilogue: C layout -> row = base + j + 8*hi, col = base + lm ----
#pragma unroll
  for (int mt = 0; mt < 4; mt++) {
#pragma unroll
    for (int nt = 0; nt < 2; nt++) {
      int col = bn + wn + nt * 16 + lm;
#pragma unroll
      for (int j = 0; j < 8; j++) {
        int row = bm + wm + mt * 16 + 8 * hi + j;
        if (row >= M || col >= N) continue;
        float v = acc[mt][nt][j];
        switch (mode) {
          case EPI_F32:
            out0[(size_t)row * N + col] = v;
            break;
          case EPI_SPLIT_SILU:   // in_proj: [0,DI)->xi f32, [DI,2DI)->silu(z) bf16
            if (col < DI) out0[(size_t)row * DI + col] = v;
            else out1[(size_t)row * DI + (col - DI)] =
                     (bh)(v / (1.f + __expf(-v)));
            break;
          case EPI_BF16:
            out1[(size_t)row * N + col] = (bh)v;
            break;
          case EPI_SOFTPLUS: {
            float u  = v + bias[col];
            float sp = (u > 20.f) ? u : log1pf(__expf(u));
            out0[(size_t)row * N + col] = sp;
          } break;
          case EPI_RESID:
            out0[(size_t)row * N + col] = v + resid[(size_t)row * N + col];
            break;
        }
      }
    }
  }
}

// ---------------------- depthwise causal conv + SiLU -----------------------

__global__ __launch_bounds__(256) void k_dwconv(
    const float* __restrict__ xi, const float* __restrict__ w,
    const float* __restrict__ bias, float* __restrict__ xc,
    bh* __restrict__ xcbf, int rev) {
  int d = blockIdx.x * 256 + threadIdx.x;
  int t = blockIdx.y, b = blockIdx.z;
  float acc = bias[d];
#pragma unroll
  for (int j = 0; j < 4; j++) {
    int ts = t - 3 + j;
    if (ts >= 0) {
      int src = rev ? (LL - 1 - ts) : ts;
      acc += w[d * 4 + j] * xi[((size_t)b * LL + src) * DI + d];
    }
  }
  float s = acc / (1.f + __expf(-acc));
  size_t idx = ((size_t)b * LL + t) * DI + d;
  xc[idx]   = s;
  xcbf[idx] = (bh)s;
}

// ------------------------------ selective scan -----------------------------
// 4 lanes per (b,d) channel, 4 states each; y written in place over xc.

__global__ __launch_bounds__(256) void k_scan(
    const float* __restrict__ delta, float* __restrict__ xc_y,
    const bh* __restrict__ xdbl, const float* __restrict__ A_log,
    const float* __restrict__ Dp, const bh* __restrict__ z_silu, int rev) {
  int g = blockIdx.x * 256 + threadIdx.x;
  int sub = g & 3;
  int dl  = g >> 2;
  int d   = dl & (DI - 1);
  int b   = dl >> 11;
  if (b >= BB) return;
  float Ar[4], h[4];
#pragma unroll
  for (int i = 0; i < 4; i++) {
    int n = sub * 4 + i;
    Ar[i] = -__expf(A_log[d * DS + n]);
    h[i]  = 0.f;
  }
  float Dd = Dp[d];
  for (int t = 0; t < LL; t++) {
    size_t idx = ((size_t)b * LL + t) * DI + d;
    float dlt = delta[idx];
    float x   = xc_y[idx];
    float dx  = dlt * x;
    size_t bc = ((size_t)b * LL + t) * XD;
    float yp = 0.f;
#pragma unroll
    for (int i = 0; i < 4; i++) {
      int n = sub * 4 + i;
      float Bn = (float)xdbl[bc + DR + n];
      float Cn = (float)xdbl[bc + DR + DS + n];
      h[i] = __expf(dlt * Ar[i]) * h[i] + dx * Bn;
      yp += h[i] * Cn;
    }
    yp += __shfl_xor(yp, 1);
    yp += __shfl_xor(yp, 2);
    if (sub == 0) {
      int zt = rev ? (LL - 1 - t) : t;
      float gz = (float)z_silu[((size_t)b * LL + zt) * DI + d];
      xc_y[idx] = (yp + Dd * x) * gz;
    }
  }
}

// --------------------------- combine directions ----------------------------

__global__ __launch_bounds__(256) void k_combine(
    const float* __restrict__ yf, const float* __restrict__ yb,
    bh* __restrict__ ybf) {
  size_t g = (size_t)blockIdx.x * 256 + threadIdx.x;
  if (g >= (size_t)ROWS * DI) return;
  int d    = (int)(g & (DI - 1));
  size_t r = g >> 11;
  int t = (int)(r & (LL - 1));
  int b = (int)(r >> 12);
  size_t bi = ((size_t)b * LL + (LL - 1 - t)) * DI + d;
  ybf[g] = (bh)(0.5f * (yf[g] + yb[bi]));
}

// ------------------------------- launcher ----------------------------------

extern "C" void kernel_launch(void* const* d_in, const int* in_sizes, int n_in,
                              void* d_out, int out_size, void* d_ws,
                              size_t ws_size, hipStream_t stream) {
  const float* x        = (const float*)d_in[0];
  const float* norm_w   = (const float*)d_in[1];
  const float* in_proj  = (const float*)d_in[2];
  const float* out_proj = (const float*)d_in[3];
  const float* conv_w_f = (const float*)d_in[4];
  const float* conv_b_f = (const float*)d_in[5];
  const float* xproj_f  = (const float*)d_in[6];
  const float* dtproj_f = (const float*)d_in[7];
  const float* dtbias_f = (const float*)d_in[8];
  const float* A_log_f  = (const float*)d_in[9];
  const float* D_f      = (const float*)d_in[10];
  const float* conv_w_b = (const float*)d_in[11];
  const float* conv_b_b = (const float*)d_in[12];
  const float* xproj_b  = (const float*)d_in[13];
  const float* dtproj_b = (const float*)d_in[14];
  const float* dtbias_b = (const float*)d_in[15];
  const float* A_log_b  = (const float*)d_in[16];
  const float* D_b      = (const float*)d_in[17];
  float* out = (float*)d_out;
  (void)in_sizes; (void)n_in; (void)out_size; (void)ws_size;

  char* ws = (char*)d_ws;
  size_t off = 0;
  auto alloc = [&](size_t bytes) -> void* {
    void* p = ws + off;
    off += (bytes + 255) & ~(size_t)255;
    return p;
  };

  bh* xn_bf      = (bh*)alloc((size_t)ROWS * DM * 2);
  bh* w_in       = (bh*)alloc((size_t)2 * DI * DM * 2);
  bh* w_out      = (bh*)alloc((size_t)DM * DI * 2);
  bh* w_xf       = (bh*)alloc((size_t)XD * DI * 2);
  bh* w_xb       = (bh*)alloc((size_t)XD * DI * 2);
  bh* w_dtf      = (bh*)alloc((size_t)DI * DR * 2);
  bh* w_dtb      = (bh*)alloc((size_t)DI * DR * 2);
  float* xi      = (float*)alloc((size_t)ROWS * DI * 4);  // reused as delta_f
  bh* z_silu     = (bh*)alloc((size_t)ROWS * DI * 2);
  float* xc_f    = (float*)alloc((size_t)ROWS * DI * 4);  // becomes y_f in place
  bh* xcbf_f     = (bh*)alloc((size_t)ROWS * DI * 2);
  float* xc_b    = (float*)alloc((size_t)ROWS * DI * 4);  // becomes y_b in place
  bh* xcbf_b     = (bh*)alloc((size_t)ROWS * DI * 2);
  bh* xdbl_f     = (bh*)alloc((size_t)ROWS * XD * 2);
  bh* xdbl_b     = (bh*)alloc((size_t)ROWS * XD * 2);
  float* delta_b = (float*)alloc((size_t)ROWS * DI * 4);
  bh* y_bf       = (bh*)alloc((size_t)ROWS * DI * 2);
  float* delta_f = xi;

  auto cvt = [&](const float* s, bh* dnew, int n) {
    k_f2bf<<<(n + 255) / 256, 256, 0, stream>>>(s, dnew, n);
  };
  cvt(in_proj,  w_in,  2 * DI * DM);
  cvt(out_proj, w_out, DM * DI);
  cvt(xproj_f,  w_xf,  XD * DI);
  cvt(xproj_b,  w_xb,  XD * DI);
  cvt(dtproj_f, w_dtf, DI * DR);
  cvt(dtproj_b, w_dtb, DI * DR);

  k_rmsnorm<<<ROWS, 256, 0, stream>>>(x, norm_w, xn_bf);

  auto gemm = [&](const bh* A, int lda, const bh* W, int M, int N, int K,
                  int mode, float* o0, bh* o1, const float* bias,
                  const float* resid) {
    dim3 grid((N + TN - 1) / TN, M / TM);
    k_gemm<<<grid, 256, 0, stream>>>(A, lda, W, M, N, K, mode, o0, o1, bias,
                                     resid);
  };

  // in_proj: xn @ Wᵀ -> xi (f32) and pre-gated silu(z) (bf16)
  gemm(xn_bf, DM, w_in, ROWS, 2 * DI, DM, EPI_SPLIT_SILU, xi, z_silu, nullptr,
       nullptr);

  // causal depthwise conv + SiLU (both time directions)
  dim3 cgrid(DI / 256, LL, BB);
  k_dwconv<<<cgrid, 256, 0, stream>>>(xi, conv_w_f, conv_b_f, xc_f, xcbf_f, 0);
  k_dwconv<<<cgrid, 256, 0, stream>>>(xi, conv_w_b, conv_b_b, xc_b, xcbf_b, 1);

  // forward branch
  gemm(xcbf_f, DI, w_xf, ROWS, XD, DI, EPI_BF16, nullptr, xdbl_f, nullptr,
       nullptr);
  gemm(xdbl_f, XD, w_dtf, ROWS, DI, DR, EPI_SOFTPLUS, delta_f, nullptr,
       dtbias_f, nullptr);
  k_scan<<<(BB * DI * 4) / 256, 256, 0, stream>>>(delta_f, xc_f, xdbl_f,
                                                  A_log_f, D_f, z_silu, 0);

  // backward branch (operates in reversed-time domain)
  gemm(xcbf_b, DI, w_xb, ROWS, XD, DI, EPI_BF16, nullptr, xdbl_b, nullptr,
       nullptr);
  gemm(xdbl_b, XD, w_dtb, ROWS, DI, DR, EPI_SOFTPLUS, delta_b, nullptr,
       dtbias_b, nullptr);
  k_scan<<<(BB * DI * 4) / 256, 256, 0, stream>>>(delta_b, xc_b, xdbl_b,
                                                  A_log_b, D_b, z_silu, 1);

  // y = 0.5*y_f + 0.5*reverse(y_b) -> bf16
  size_t tot = (size_t)ROWS * DI;
  k_combine<<<(unsigned)((tot + 255) / 256), 256, 0, stream>>>(xc_f, xc_b,
                                                               y_bf);

  // out_proj + residual -> d_out
  gemm(y_bf, DI, w_out, ROWS, DM, DI, EPI_RESID, out, nullptr, nullptr, x);
}
